// MetaLayer_t_19292993094376
// MI455X (gfx1250) — matrix operand order repaired and checked by reading, not compiled, not run on hardware
//
#include <hip/hip_runtime.h>

// ---------------------------------------------------------------------------
// MetaLayer (GNN edge+node MLP) for MI455X / gfx1250, wave32, f16 WMMA.
//
// Roofline: ~55 GFLOP vs ~1.1 GB of HBM traffic -> memory bound (~50us floor
// at 23.3 TB/s).  GEMMs use v_wmma_f32_16x16x32_f16 (f32 accumulate).  The
// edge kernel fuses gather -> GEMM -> relu -> GEMM -> scatter, so the
// [800k,192] concat matrix (615 MB) never touches HBM.  Each wave now carries
// TWO 16-edge A-tiles so every weight B-fragment load from L2 feeds two
// WMMAs, halving L2 fragment traffic (3.2 GB -> 1.6 GB).
// ---------------------------------------------------------------------------

typedef __attribute__((ext_vector_type(16))) _Float16 v16h;
typedef __attribute__((ext_vector_type(8)))  _Float16 v8h;
typedef __attribute__((ext_vector_type(4)))  _Float16 v4h;
typedef __attribute__((ext_vector_type(2)))  _Float16 v2h;
typedef __attribute__((ext_vector_type(8)))  float    v8f;

#define NN   50000
#define NE   800000
#define FN   64
#define FEA  64
#define HID  128
#define KIN  192   // 2*FN + FEA

// Build a 16x32 f16 A-fragment from an LDS row (row-major), contiguous K:
// lanes 0-15: M=lane,    halfs 0..15 -> K=32k+0..15
// lanes16-31: M=lane-16, halfs 0..15 -> K=32k+16..31
__device__ __forceinline__ v16h a_frag(const _Float16* row, int ktile, int lane) {
  const int kb = (lane >> 4) << 4;                    // 0 or 16 halfs
  const v8h lo = *(const v8h*)(row + ktile * 32 + kb);
  const v8h hi = *(const v8h*)(row + ktile * 32 + kb + 8);
  return __builtin_shufflevector(lo, hi, 0,1,2,3,4,5,6,7,8,9,10,11,12,13,14,15);
}

// Zero-fill scratch (agg + cnt).
__global__ void zero_kernel(float* __restrict__ p, long n) {
  long i = (long)blockIdx.x * blockDim.x + threadIdx.x;
  if (i < n) p[i] = 0.0f;
}

// Repack a row-major f32 weight [K][N] into f16 WMMA B-fragments:
// frag (kt,nt): lane L holds B[32kt + (L>=16?16:0) + {2r,2r+1}][16nt + (L&15)]
// stored as 16 consecutive halfs per lane -> one 32B coalesced load at use.
__global__ void pack_w_kernel(const float* __restrict__ W, _Float16* __restrict__ P,
                              int K, int N) {
  const int t = blockIdx.x * blockDim.x + threadIdx.x;
  const int nT = N >> 4;
  const int frags = (K >> 5) * nT;
  if (t >= frags * 32) return;
  const int frag = t >> 5, lane = t & 31;
  const int kt = frag / nT, nt = frag % nT;
  const int n  = nt * 16 + (lane & 15);
  const int kb = kt * 32 + ((lane >> 4) << 4);
  _Float16* dst = P + ((size_t)frag * 32 + lane) * 16;
#pragma unroll
  for (int r = 0; r < 8; ++r) {
    dst[2 * r]     = (_Float16)W[(size_t)(kb + 2 * r)     * N + n];
    dst[2 * r + 1] = (_Float16)W[(size_t)(kb + 2 * r + 1) * N + n];
  }
}

// ------------- Edge MLP: one wave32 per 32-edge (2 x 16) tile -------------
__global__ __launch_bounds__(64) void edge_mlp_kernel(
    const float* __restrict__ x, const int* __restrict__ ei,
    const float* __restrict__ ea,
    const _Float16* __restrict__ We1p, const float* __restrict__ be1,
    const _Float16* __restrict__ We2p, const float* __restrict__ be2,
    float* __restrict__ out_edge, float* __restrict__ agg,
    float* __restrict__ cnt) {
  __shared__ __align__(16) _Float16 s_in[2][32][KIN];   // 24 KB
  __shared__ __align__(16) _Float16 s_h [2][32][HID];   // 16 KB
  __shared__ int s_dst[2][32];

  const int  wave = threadIdx.x >> 5;          // 0..1
  const int  lane = threadIdx.x & 31;
  const long base = ((long)blockIdx.x * 2 + wave) * 32;   // 32 edges / wave
  if (base >= NE) return;

  // ---- gather: edge_in = [x[src] | x[dst] | edge_attr]  (f32 -> f16 LDS) ----
  for (int e = 0; e < 32; ++e) {
    const long eg  = base + e;
    const int  src = ei[eg];
    const int  dst = ei[(long)NE + eg];
    if (lane == 0) s_dst[wave][e] = dst;
    const float2 a = *(const float2*)(x  + (long)src * FN  + 2 * lane);
    const float2 b = *(const float2*)(x  + (long)dst * FN  + 2 * lane);
    const float2 c = *(const float2*)(ea + eg        * FEA + 2 * lane);
    v2h va = {(_Float16)a.x, (_Float16)a.y};
    v2h vb = {(_Float16)b.x, (_Float16)b.y};
    v2h vc = {(_Float16)c.x, (_Float16)c.y};
    *(v2h*)&s_in[wave][e][      2 * lane] = va;
    *(v2h*)&s_in[wave][e][ 64 + 2 * lane] = vb;
    *(v2h*)&s_in[wave][e][128 + 2 * lane] = vc;
  }
  // in-degree counts for the mean aggregation (one edge per lane, no diverge)
  atomicAdd(&cnt[s_dst[wave][lane]], 1.0f);

  const int col0 = lane & 15;
  const int mrow = (lane < 16) ? 0 : 8;
  const _Float16* arow0 = &s_in[wave][col0][0];        // tile 0: edges 0..15
  const _Float16* arow1 = &s_in[wave][16 + col0][0];   // tile 1: edges 16..31

  // ---- GEMM1: 2 x ([16,192] @ [192,128]), each B frag feeds two WMMAs ----
  v8f acc0[8] = {}, acc1[8] = {};
  const v16h* W1 = (const v16h*)We1p;
#pragma unroll
  for (int k = 0; k < 6; ++k) {
    const v16h a0 = a_frag(arow0, k, lane);
    const v16h a1 = a_frag(arow1, k, lane);
#pragma unroll
    for (int n = 0; n < 8; ++n) {
      const v16h b = W1[(size_t)(k * 8 + n) * 32 + lane];
      acc0[n] = __builtin_amdgcn_wmma_f32_16x16x32_f16(
          false, a0, false, b, (short)0, acc0[n], false, false);
      acc1[n] = __builtin_amdgcn_wmma_f32_16x16x32_f16(
          false, a1, false, b, (short)0, acc1[n], false, false);
    }
  }
  // bias + relu -> f16 hidden in LDS
#pragma unroll
  for (int n = 0; n < 8; ++n) {
    const float b1 = be1[16 * n + col0];
#pragma unroll
    for (int r = 0; r < 8; ++r) {
      float v0 = acc0[n][r] + b1;
      float v1 = acc1[n][r] + b1;
      v0 = v0 > 0.0f ? v0 : 0.0f;
      v1 = v1 > 0.0f ? v1 : 0.0f;
      s_h[wave][mrow + r][16 * n + col0]      = (_Float16)v0;
      s_h[wave][16 + mrow + r][16 * n + col0] = (_Float16)v1;
    }
  }

  // ---- GEMM2: 2 x ([16,128] @ [128,64]) ----
  v8f c20[4] = {}, c21[4] = {};
  const _Float16* hrow0 = &s_h[wave][col0][0];
  const _Float16* hrow1 = &s_h[wave][16 + col0][0];
  const v16h* W2 = (const v16h*)We2p;
#pragma unroll
  for (int k = 0; k < 4; ++k) {
    const v16h a0 = a_frag(hrow0, k, lane);
    const v16h a1 = a_frag(hrow1, k, lane);
#pragma unroll
    for (int n = 0; n < 4; ++n) {
      const v16h b = W2[(size_t)(k * 4 + n) * 32 + lane];
      c20[n] = __builtin_amdgcn_wmma_f32_16x16x32_f16(
          false, a0, false, b, (short)0, c20[n], false, false);
      c21[n] = __builtin_amdgcn_wmma_f32_16x16x32_f16(
          false, a1, false, b, (short)0, c21[n], false, false);
    }
  }
  // bias, write new edge_attr, scatter-add into agg[dst]
#pragma unroll
  for (int n = 0; n < 4; ++n) {
    const float b2 = be2[16 * n + col0];
#pragma unroll
    for (int r = 0; r < 8; ++r) {
      const int M0 = mrow + r;
      const int M1 = 16 + mrow + r;
      const float v0 = c20[n][r] + b2;
      const float v1 = c21[n][r] + b2;
      out_edge[(base + M0) * FEA + 16 * n + col0] = v0;
      out_edge[(base + M1) * FEA + 16 * n + col0] = v1;
      atomicAdd(&agg[(long)s_dst[wave][M0] * FEA + 16 * n + col0], v0);
      atomicAdd(&agg[(long)s_dst[wave][M1] * FEA + 16 * n + col0], v1);
    }
  }
}

// ---------------- Node MLP: one wave32 per 16-node tile ----------------
__global__ __launch_bounds__(128) void node_mlp_kernel(
    const float* __restrict__ x, const float* __restrict__ agg,
    const float* __restrict__ cnt,
    const _Float16* __restrict__ Wn1p, const float* __restrict__ bn1,
    const _Float16* __restrict__ Wn2p, const float* __restrict__ bn2,
    float* __restrict__ out_x) {
  __shared__ __align__(16) _Float16 s_in[4][16][HID];   // 16 KB
  __shared__ __align__(16) _Float16 s_h [4][16][HID];   // 16 KB

  const int  wave = threadIdx.x >> 5;
  const int  lane = threadIdx.x & 31;
  const long base = ((long)blockIdx.x * 4 + wave) * 16;
  if (base >= NN) return;

  // ---- build node_in = [x | agg/max(cnt,1)] in LDS (f16) ----
  for (int nn = 0; nn < 16; ++nn) {
    const long node = base + nn;
    if (lane < 16) {
      const float4 a = *(const float4*)(x + node * FN + 4 * lane);
      v4h v = {(_Float16)a.x, (_Float16)a.y, (_Float16)a.z, (_Float16)a.w};
      *(v4h*)&s_in[wave][nn][4 * lane] = v;
    } else {
      const int   l   = lane - 16;
      const float c   = cnt[node];
      const float inv = 1.0f / (c > 1.0f ? c : 1.0f);
      const float4 a  = *(const float4*)(agg + node * FEA + 4 * l);
      v4h v = {(_Float16)(a.x * inv), (_Float16)(a.y * inv),
               (_Float16)(a.z * inv), (_Float16)(a.w * inv)};
      *(v4h*)&s_in[wave][nn][64 + 4 * l] = v;
    }
  }

  const int col0 = lane & 15;
  const int mrow = (lane < 16) ? 0 : 8;
  const _Float16* arow = &s_in[wave][col0][0];

  // ---- GEMM1: [16,128] @ [128,128] ----
  v8f acc[8] = {};
  const v16h* W1 = (const v16h*)Wn1p;
#pragma unroll
  for (int k = 0; k < 4; ++k) {
    const v16h a = a_frag(arow, k, lane);
#pragma unroll
    for (int n = 0; n < 8; ++n) {
      const v16h b = W1[(size_t)(k * 8 + n) * 32 + lane];
      acc[n] = __builtin_amdgcn_wmma_f32_16x16x32_f16(
          false, a, false, b, (short)0, acc[n], false, false);
    }
  }
#pragma unroll
  for (int n = 0; n < 8; ++n) {
    const float b1 = bn1[16 * n + col0];
#pragma unroll
    for (int r = 0; r < 8; ++r) {
      float v = acc[n][r] + b1;
      v = v > 0.0f ? v : 0.0f;
      s_h[wave][mrow + r][16 * n + col0] = (_Float16)v;
    }
  }

  // ---- GEMM2: [16,128] @ [128,64] ----
  v8f c2[4] = {};
  const _Float16* hrow = &s_h[wave][col0][0];
  const v16h* W2 = (const v16h*)Wn2p;
#pragma unroll
  for (int k = 0; k < 4; ++k) {
    const v16h a = a_frag(hrow, k, lane);
#pragma unroll
    for (int n = 0; n < 4; ++n) {
      const v16h b = W2[(size_t)(k * 4 + n) * 32 + lane];
      c2[n] = __builtin_amdgcn_wmma_f32_16x16x32_f16(
          false, a, false, b, (short)0, c2[n], false, false);
    }
  }
#pragma unroll
  for (int n = 0; n < 4; ++n) {
    const float b2 = bn2[16 * n + col0];
#pragma unroll
    for (int r = 0; r < 8; ++r) {
      const long row = base + mrow + r;
      out_x[row * FN + 16 * n + col0] = c2[n][r] + b2;
    }
  }
}

// ---------------------------------------------------------------------------
extern "C" void kernel_launch(void* const* d_in, const int* in_sizes, int n_in,
                              void* d_out, int out_size, void* d_ws, size_t ws_size,
                              hipStream_t stream) {
  const float* x   = (const float*)d_in[0];
  const int*   ei  = (const int*)  d_in[1];   // edge_index [2, NE]
  const float* ea  = (const float*)d_in[2];
  const float* We1 = (const float*)d_in[3];
  const float* be1 = (const float*)d_in[4];
  const float* We2 = (const float*)d_in[5];
  const float* be2 = (const float*)d_in[6];
  const float* Wn1 = (const float*)d_in[7];
  const float* bn1 = (const float*)d_in[8];
  const float* Wn2 = (const float*)d_in[9];
  const float* bn2 = (const float*)d_in[10];

  float* out_x = (float*)d_out;                 // tuple: x_new first ...
  float* out_e = out_x + (size_t)NN * FN;       // ... then edge_attr_new

  // workspace: agg | cnt | packed f16 weights
  char*  ws = (char*)d_ws;
  size_t o  = 0;
  float* agg = (float*)(ws + o); o += (size_t)NN * FEA * 4;
  float* cnt = (float*)(ws + o); o += (size_t)NN * 4;
  o = (o + 255) & ~(size_t)255;
  _Float16* We1p = (_Float16*)(ws + o); o += (size_t)KIN * HID * 2;
  _Float16* We2p = (_Float16*)(ws + o); o += (size_t)HID * FEA * 2;
  _Float16* Wn1p = (_Float16*)(ws + o); o += (size_t)HID * HID * 2;
  _Float16* Wn2p = (_Float16*)(ws + o); o += (size_t)HID * FN  * 2;

  // 1) zero scratch (agg and cnt are contiguous)
  {
    const long n = (long)NN * FEA + NN;
    zero_kernel<<<(unsigned)((n + 255) / 256), 256, 0, stream>>>(agg, n);
  }
  // 2) repack weights into WMMA fragment layout (f32 -> f16)
  pack_w_kernel<<<6, 256, 0, stream>>>(We1, We1p, KIN, HID);  // 48 frags
  pack_w_kernel<<<2, 256, 0, stream>>>(We2, We2p, HID, FEA);  // 16 frags
  pack_w_kernel<<<4, 256, 0, stream>>>(Wn1, Wn1p, HID, HID);  // 32 frags
  pack_w_kernel<<<2, 256, 0, stream>>>(Wn2, Wn2p, HID, FN);   // 16 frags
  // 3) fused edge MLP + scatter aggregation (64 edges / block of 2 waves)
  edge_mlp_kernel<<<NE / 64, 64, 0, stream>>>(x, ei, ea, We1p, be1, We2p, be2,
                                              out_e, agg, cnt);
  // 4) fused node MLP (64 nodes / block)
  node_mlp_kernel<<<(NN / 16 + 3) / 4, 128, 0, stream>>>(x, agg, cnt, Wn1p, bn1,
                                                         Wn2p, bn2, out_x);
}
